// lstm_encoder_decoder_2628519985747
// MI455X (gfx1250) — compile-verified, matrix operand order checked by reading, not run on hardware
//
#include <hip/hip_runtime.h>

// ---------------------------------------------------------------------------
// LSTM encoder-decoder (B=512, H=512, enc T=512, dec T=128) for gfx1250.
// Persistent cooperative kernel: one wave owns one 16x16 (batch x hidden)
// tile and computes all four gates with v_wmma_f32_16x16x32_bf16.
// h state double-buffered in bf16; weights converted to bf16 once per call.
// Device-wide sync via monotonic-ticket atomic barrier (no launch per step).
// ---------------------------------------------------------------------------

typedef __attribute__((ext_vector_type(16))) __bf16 v16bf;
typedef __attribute__((ext_vector_type(8)))  float  v8f;
typedef __attribute__((ext_vector_type(2)))  __bf16 v2bf;

#define HID   512
#define BATCH 512
#define TENC  512
#define NBLK  128
#define NTHR  256

__device__ __forceinline__ float sigf(float x) { return 1.0f / (1.0f + __expf(-x)); }

// Monotonic grid barrier: no counter reset needed; counter is zeroed by a
// hipMemsetAsync before the kernel (capture-safe). ~770 epochs * 128 << 2^32.
__device__ __forceinline__ void grid_sync(unsigned* cnt) {
  __syncthreads();
  if (threadIdx.x == 0) {
    __threadfence();
    const unsigned nb  = gridDim.x;
    const unsigned tk  = atomicAdd(cnt, 1u);
    const unsigned tgt = (tk / nb + 1u) * nb;
    while (__hip_atomic_load(cnt, __ATOMIC_ACQUIRE, __HIP_MEMORY_SCOPE_AGENT) < tgt)
      __builtin_amdgcn_s_sleep(1);
  }
  __syncthreads();
}

__global__ __launch_bounds__(NTHR) void lstm_s2s_persistent(
    const float* __restrict__ inputs,
    const float* __restrict__ encWih, const float* __restrict__ encWhh, const float* __restrict__ encB,
    const float* __restrict__ decWih, const float* __restrict__ decWhh, const float* __restrict__ decB,
    const float* __restrict__ linW,   const float* __restrict__ linB,
    float* __restrict__ out, int tlen,
    unsigned* cnt,
    __bf16* encWbf, __bf16* decWbf,
    __bf16* hbf0, __bf16* hbf1,
    float* hf32, float* cst, float* xdec)
{
  __shared__ __bf16 hs[16 * HID];   // 16 batch rows of h, shared by 8 waves

  const int tid  = threadIdx.x;
  const int lane = tid & 31;
  const int wv   = tid >> 5;
  const int gw   = blockIdx.x * (NTHR / 32) + wv;  // 0..1023
  const int mt   = gw >> 5;                        // batch tile; same for all waves in block
  const int jt   = gw & 31;                        // hidden tile
  const int M0   = mt * 16;
  const int J0   = jt * 16;
  const int half = lane >> 4;                      // ISA half-lane split
  const int ln16 = lane & 15;

  // ---- phase 0: bf16 weight conversion + state init ----
  {
    const int gtid = blockIdx.x * NTHR + tid;
    const int nthr = NBLK * NTHR;
    for (int i = gtid; i < 4 * HID * HID; i += nthr) {
      encWbf[i] = (__bf16)encWhh[i];
      decWbf[i] = (__bf16)decWhh[i];
    }
    for (int i = gtid; i < BATCH * HID; i += nthr) {
      hbf0[i] = (__bf16)0.0f;
      hbf1[i] = (__bf16)0.0f;
      cst[i]  = 0.0f;
      hf32[i] = 0.0f;
    }
    if (gtid < BATCH) xdec[gtid] = inputs[(size_t)gtid * TENC + (TENC - 1)];
  }
  grid_sync(cnt);

  const int kidx[8] = {0, 2, 4, 6, 16, 18, 20, 22};  // packed-K order per ISA 7.12.2

  for (int s = 0; s < TENC + tlen; ++s) {
    const bool    enc  = (s < TENC);
    const __bf16* wbf  = enc ? encWbf : decWbf;
    const float*  bias = enc ? encB   : decB;
    const float*  wih  = enc ? encWih : decWih;
    const __bf16* hrd  = (s & 1) ? hbf1 : hbf0;
    __bf16*       hwr  = (s & 1) ? hbf0 : hbf1;

    // stage this block's 16 rows of h(t) into LDS (dword copies)
    {
      const unsigned* src = (const unsigned*)(hrd + (size_t)M0 * HID);
      unsigned*       dst = (unsigned*)hs;
      for (int i = tid; i < 16 * HID / 2; i += NTHR) dst[i] = src[i];
    }
    __syncthreads();

    v8f acc[4] = {};   // i, f, g, o accumulator tiles

    const __bf16* wr0  = wbf + (size_t)(0 * HID + J0 + ln16) * HID;
    const __bf16* wr1  = wbf + (size_t)(1 * HID + J0 + ln16) * HID;
    const __bf16* wr2  = wbf + (size_t)(2 * HID + J0 + ln16) * HID;
    const __bf16* wr3  = wbf + (size_t)(3 * HID + J0 + ln16) * HID;
    const __bf16* arow = hs + ln16 * HID;

    #pragma unroll 2
    for (int K0 = 0; K0 < HID; K0 += 32) {
      v16bf a;
      #pragma unroll
      for (int v = 0; v < 8; ++v) {
        v2bf p = *(const v2bf*)(arow + K0 + kidx[v] + 8 * half);
        a[2 * v] = p[0]; a[2 * v + 1] = p[1];
      }
      const __bf16* wr[4] = {wr0, wr1, wr2, wr3};
      #pragma unroll
      for (int gi = 0; gi < 4; ++gi) {
        v16bf bm;
        #pragma unroll
        for (int v = 0; v < 8; ++v) {
          v2bf p = *(const v2bf*)(wr[gi] + K0 + kidx[v] + 8 * half);
          bm[2 * v] = p[0]; bm[2 * v + 1] = p[1];
        }
        acc[gi] = __builtin_amdgcn_wmma_f32_16x16x32_bf16(
            false, a, false, bm, (short)0, acc[gi], false, false);
      }
    }

    // epilogue: z = wmma + b + x*W_ih ; gates ; update c, h
    const int   n   = J0 + ln16;
    const float bb0 = bias[0 * HID + n], bb1 = bias[1 * HID + n];
    const float bb2 = bias[2 * HID + n], bb3 = bias[3 * HID + n];
    const float ww0 = wih[0 * HID + n], ww1 = wih[1 * HID + n];
    const float ww2 = wih[2 * HID + n], ww3 = wih[3 * HID + n];

    #pragma unroll
    for (int r = 0; r < 8; ++r) {
      const int    b  = M0 + r + 8 * half;   // C/D row layout: r + 8*(lane>=16)
      const float  xv = enc ? inputs[(size_t)b * TENC + s] : xdec[b];
      const float  zi = acc[0][r] + bb0 + xv * ww0;
      const float  zf = acc[1][r] + bb1 + xv * ww1;
      const float  zg = acc[2][r] + bb2 + xv * ww2;
      const float  zo = acc[3][r] + bb3 + xv * ww3;
      const size_t ix = (size_t)b * HID + n;
      const float  cn = sigf(zf) * cst[ix] + sigf(zi) * tanhf(zg);
      const float  hn = sigf(zo) * tanhf(cn);
      cst[ix]  = cn;
      hf32[ix] = hn;
      hwr[ix]  = (__bf16)hn;
    }

    grid_sync(cnt);

    if (!enc) {
      // decoder head: out[b,t] = h . linW + linB, feed back as next x
      const int t = s - TENC;
      if (gw < BATCH) {
        float p = 0.0f;
        const float* hv = hf32 + (size_t)gw * HID;
        #pragma unroll 4
        for (int i = lane; i < HID; i += 32) p += hv[i] * linW[i];
        #pragma unroll
        for (int off = 16; off > 0; off >>= 1) p += __shfl_xor(p, off, 32);
        if (lane == 0) {
          const float o = p + linB[0];
          out[(size_t)gw * tlen + t] = o;
          xdec[gw] = o;
        }
      }
      grid_sync(cnt);
    }
  }
}

extern "C" void kernel_launch(void* const* d_in, const int* in_sizes, int n_in,
                              void* d_out, int out_size, void* d_ws, size_t ws_size,
                              hipStream_t stream) {
  (void)in_sizes; (void)n_in; (void)ws_size;
  const float* inputs = (const float*)d_in[0];
  const float* encWih = (const float*)d_in[2];
  const float* encWhh = (const float*)d_in[3];
  const float* encB   = (const float*)d_in[4];
  const float* decWih = (const float*)d_in[5];
  const float* decWhh = (const float*)d_in[6];
  const float* decB   = (const float*)d_in[7];
  const float* linW   = (const float*)d_in[8];
  const float* linB   = (const float*)d_in[9];
  float*       out    = (float*)d_out;
  const int    tlen   = out_size / BATCH;   // 128

  // workspace carve-up (~7.4 MB), 256B aligned
  char*    ws     = (char*)d_ws;
  size_t   off    = 0;
  unsigned* cnt   = (unsigned*)(ws + off); off += 256;
  __bf16* encWbf  = (__bf16*)(ws + off);   off += (size_t)4 * HID * HID * 2;
  __bf16* decWbf  = (__bf16*)(ws + off);   off += (size_t)4 * HID * HID * 2;
  __bf16* hbf0    = (__bf16*)(ws + off);   off += (size_t)BATCH * HID * 2;
  __bf16* hbf1    = (__bf16*)(ws + off);   off += (size_t)BATCH * HID * 2;
  float*  hf32    = (float*)(ws + off);    off += (size_t)BATCH * HID * 4;
  float*  cst     = (float*)(ws + off);    off += (size_t)BATCH * HID * 4;
  float*  xdec    = (float*)(ws + off);    off += (size_t)BATCH * 4;

  // zero the barrier counter every call (deterministic, capture-safe)
  hipMemsetAsync(d_ws, 0, 256, stream);

  lstm_s2s_persistent<<<NBLK, NTHR, 0, stream>>>(
      inputs, encWih, encWhh, encB, decWih, decWhh, decB, linW, linB,
      out, tlen, cnt, encWbf, decWbf, hbf0, hbf1, hf32, cst, xdec);
}
